// CausalAttentionV1_90056874262677
// MI455X (gfx1250) — matrix operand order repaired and checked by reading, not compiled, not run on hardware
//
#include <hip/hip_runtime.h>
#include <hip/hip_bf16.h>
#include <stdint.h>

typedef __bf16 bf16_t;
typedef __attribute__((ext_vector_type(16))) __bf16 v16bf;
typedef __attribute__((ext_vector_type(8)))  __bf16 v8bf;
typedef __attribute__((ext_vector_type(4)))  __bf16 v4bf;
typedef __attribute__((ext_vector_type(8)))  float  v8f;
typedef __attribute__((ext_vector_type(4)))  float  v4f;
typedef __attribute__((ext_vector_type(4)))  uint32_t v4u;
typedef __attribute__((ext_vector_type(8)))  uint32_t v8u;

#define BATCH   4
#define SEQ     2048
#define DMODEL  1024
#define ROWS    (BATCH * SEQ)     // 8192
#define BK      32
#define LDT     40                // LDS row stride in bf16 elems (80B: conflict-free, 16B aligned)

// ---- WMMA fragment loaders (CDNA5 16-bit layouts, wave32) -------------------
// A 16x32: lane L<16 -> M=L, K chunks [0..7],[16..23]; lane L>=16 -> M=L-16, [8..15],[24..31]
static __device__ __forceinline__ v16bf load_frag_a(const bf16_t* __restrict__ base,
                                                    int ld, int lane) {
  const int m  = lane & 15;
  const int k0 = (lane & 16) ? 8 : 0;
  const bf16_t* p = base + (size_t)m * ld + k0;
  v8bf lo = *(const v8bf*)(p);
  v8bf hi = *(const v8bf*)(p + 16);
  v16bf r;
#pragma unroll
  for (int i = 0; i < 8; ++i) { r[i] = lo[i]; r[i + 8] = hi[i]; }
  return r;
}
// B 32x16: lane L<16 -> N=L, K chunks [0..7],[16..23]; lane L>=16 -> N=L-16, [8..15],[24..31]
// `base` must point at storage where row n holds contiguous K (i.e. B^T row-major).
static __device__ __forceinline__ v16bf load_frag_b(const bf16_t* __restrict__ base,
                                                    int ld, int lane) {
  const int n  = lane & 15;
  const int k0 = (lane & 16) ? 8 : 0;
  const bf16_t* p = base + (size_t)n * ld + k0;
  v8bf lo = *(const v8bf*)(p);
  v8bf hi = *(const v8bf*)(p + 16);
  v16bf r;
#pragma unroll
  for (int i = 0; i < 8; ++i) { r[i] = lo[i]; r[i + 8] = hi[i]; }
  return r;
}

static __device__ __forceinline__ v8f wmma_bf16(v16bf a, v16bf b, v8f c) {
  return __builtin_amdgcn_wmma_f32_16x16x32_bf16(false, a, false, b,
                                                 (short)0, c, false, false);
}

// ---- Tensor Data Mover: async 2D bf16 tile (32 elems x H rows) global -> LDS.
// Pads LDS dest by 4 DWORDs (16B) every 16 DWORDs (64B) => 80B row stride = LDT.
// Issue from ONE wave only; drain with s_wait_tensorcnt.
static __device__ __forceinline__ void tdm_load_tile_2d(uint32_t lds_off,
                                                        const bf16_t* gptr,
                                                        uint32_t tile_h,
                                                        uint32_t stride_elems,
                                                        uint32_t tensor_w,
                                                        uint32_t tensor_h) {
  const uint64_t ga = (uint64_t)(uintptr_t)gptr;
  v4u g0;
  g0[0] = 1u;                                   // count=1, user descriptor
  g0[1] = lds_off;                              // LDS byte address
  g0[2] = (uint32_t)ga;                         // global_addr[31:0]  (bits 95:64)
  g0[3] = (uint32_t)(ga >> 32) | (2u << 30);    // global_addr[56:32] | type=2
  v8u g1;
  g1[0] = (1u << 16)        // data_size = 1 -> 2 bytes
        | (1u << 20)        // pad_enable
        | (3u << 22)        // pad_interval = 3 -> every 16 DWORDs (64B)
        | (3u << 25);       // pad_amount  = 3 -> 4 DWORDs (16B)
  g1[1] = (tensor_w & 0xFFFFu) << 16;           // tensor_dim0[15:0]
  g1[2] = (tensor_w >> 16) | ((tensor_h & 0xFFFFu) << 16);  // td0 hi | td1 lo
  g1[3] = (tensor_h >> 16) | (32u << 16);       // td1 hi | tile_dim0 = 32
  g1[4] = tile_h & 0xFFFFu;                     // tile_dim1 | tile_dim2=0
  g1[5] = stride_elems;                         // tensor_dim0_stride[31:0]
  g1[6] = 0u;                                   // stride hi | dim1_stride lo
  g1[7] = 0u;
  asm volatile("tensor_load_to_lds %0, %1" :: "s"(g0), "s"(g1) : "memory");
}

// ---- Kernel 1: Y[bf16] = X[f32] @ W[f32]; block 64(M)x128(N), 4 waves,
//      double-buffered LDS, one barrier per K-step ---------------------------
__global__ __launch_bounds__(128)
void qkv_gemm_kernel(const float* __restrict__ X, const float* __restrict__ W,
                     bf16_t* __restrict__ Y, float out_scale) {
  __shared__ bf16_t sA[2][64 * LDT];            // 64 x 32, K contiguous
  __shared__ bf16_t sB[2][128 * LDT];           // B^T: 128(n) x 32(k)
  const int tid  = threadIdx.x;
  const int lane = tid & 31;
  const int wave = tid >> 5;                    // wave owns N strip wave*32
  const int m0   = blockIdx.y * 64;
  const int n0   = blockIdx.x * 128;

  v4f regA[4], regB[8];
  // prologue: stage K-step 0 into registers
#pragma unroll
  for (int i = 0; i < 4; ++i) {
    const int idx = tid + i * 128;
    const int r = idx >> 3, c4 = idx & 7;
    regA[i] = *(const v4f*)(X + (size_t)(m0 + r) * DMODEL + c4 * 4);
  }
#pragma unroll
  for (int i = 0; i < 8; ++i) {
    const int idx = tid + i * 128;
    const int r = idx >> 5, c4 = idx & 31;
    regB[i] = *(const v4f*)(W + (size_t)r * DMODEL + n0 + c4 * 4);
  }

  v8f acc[4][2] = {};
  for (int k0 = 0, step = 0; k0 < DMODEL; k0 += BK, ++step) {
    const int cur = step & 1;
    // commit staged registers (f32 -> bf16) into LDS buffer `cur`
#pragma unroll
    for (int i = 0; i < 4; ++i) {
      const int idx = tid + i * 128;
      const int r = idx >> 3, c4 = idx & 7;
      v4bf h;
      h[0] = (bf16_t)regA[i][0]; h[1] = (bf16_t)regA[i][1];
      h[2] = (bf16_t)regA[i][2]; h[3] = (bf16_t)regA[i][3];
      *(v4bf*)(&sA[cur][r * LDT + c4 * 4]) = h;
    }
#pragma unroll
    for (int i = 0; i < 8; ++i) {
      const int idx = tid + i * 128;
      const int r = idx >> 5, c4 = idx & 31;
#pragma unroll
      for (int j = 0; j < 4; ++j) sB[cur][(c4 * 4 + j) * LDT + r] = (bf16_t)regB[i][j];
    }
    __syncthreads();
    // prefetch K-step k0+BK into registers (hides under the WMMAs below)
    if (k0 + BK < DMODEL) {
#pragma unroll
      for (int i = 0; i < 4; ++i) {
        const int idx = tid + i * 128;
        const int r = idx >> 3, c4 = idx & 7;
        regA[i] = *(const v4f*)(X + (size_t)(m0 + r) * DMODEL + (k0 + BK) + c4 * 4);
      }
#pragma unroll
      for (int i = 0; i < 8; ++i) {
        const int idx = tid + i * 128;
        const int r = idx >> 5, c4 = idx & 31;
        regB[i] = *(const v4f*)(W + (size_t)(k0 + BK + r) * DMODEL + n0 + c4 * 4);
      }
    }
    v16bf af[4], bfv[2];
#pragma unroll
    for (int mi = 0; mi < 4; ++mi) af[mi] = load_frag_a(&sA[cur][(mi * 16) * LDT], LDT, lane);
#pragma unroll
    for (int ni = 0; ni < 2; ++ni)
      bfv[ni] = load_frag_b(&sB[cur][(wave * 32 + ni * 16) * LDT], LDT, lane);
#pragma unroll
    for (int mi = 0; mi < 4; ++mi)
#pragma unroll
      for (int ni = 0; ni < 2; ++ni)
        acc[mi][ni] = wmma_bf16(af[mi], bfv[ni], acc[mi][ni]);
    // no trailing barrier: next step writes the other buffer
  }

  const int rbase = m0 + ((lane & 16) ? 8 : 0);
  const int cbase = n0 + wave * 32 + (lane & 15);
#pragma unroll
  for (int mi = 0; mi < 4; ++mi)
#pragma unroll
    for (int ni = 0; ni < 2; ++ni)
#pragma unroll
      for (int v = 0; v < 8; ++v) {
        const int row = rbase + mi * 16 + v;
        const int col = cbase + ni * 16;
        Y[(size_t)row * DMODEL + col] = (bf16_t)(acc[mi][ni][v] * out_scale);
      }
}

// ---- Kernel 2: S[f32] = Qs @ K^T; block 64(q)x128(k), causal tiles only -----
__global__ __launch_bounds__(128)
void scores_kernel(const bf16_t* __restrict__ Q, const bf16_t* __restrict__ K,
                   float* __restrict__ Sc) {
  const int b   = blockIdx.z;
  const int q0  = blockIdx.y * 64;
  const int k0t = blockIdx.x * 128;
  if (k0t > q0 + 63) return;                    // fully masked tile

  const int tid  = threadIdx.x;
  const int lane = tid & 31;
  const int wave = tid >> 5;
  const int wm   = (wave >> 1) * 32;            // 32 q-rows per wave
  const int wn   = (wave & 1) * 64;             // 64 k-cols per wave
  const bf16_t* Qb = Q + (size_t)b * SEQ * DMODEL;
  const bf16_t* Kb = K + (size_t)b * SEQ * DMODEL;

  v8f acc[2][4] = {};
  for (int d = 0; d < DMODEL; d += BK) {
    if (d + BK < DMODEL) {                      // stage next K-step into caches
      __builtin_prefetch((const void*)(Qb + (size_t)(q0 + wm) * DMODEL + d + BK), 0, 0);
      __builtin_prefetch((const void*)(Kb + (size_t)(k0t + wn) * DMODEL + d + BK), 0, 0);
    }
    v16bf af[2], bfv[4];
#pragma unroll
    for (int mi = 0; mi < 2; ++mi)
      af[mi] = load_frag_a(Qb + (size_t)(q0 + wm + mi * 16) * DMODEL + d, DMODEL, lane);
    // B operand: column n == key row (k0t+wn+ni*16+n), contiguous d -> direct global load
#pragma unroll
    for (int ni = 0; ni < 4; ++ni)
      bfv[ni] = load_frag_b(Kb + (size_t)(k0t + wn + ni * 16) * DMODEL + d, DMODEL, lane);
#pragma unroll
    for (int mi = 0; mi < 2; ++mi)
#pragma unroll
      for (int ni = 0; ni < 4; ++ni)
        acc[mi][ni] = wmma_bf16(af[mi], bfv[ni], acc[mi][ni]);
  }

  float* Sb = Sc + (size_t)b * SEQ * SEQ;
  const int rbase = q0 + wm + ((lane & 16) ? 8 : 0);
  const int cbase = k0t + wn + (lane & 15);
#pragma unroll
  for (int mi = 0; mi < 2; ++mi)
#pragma unroll
    for (int ni = 0; ni < 4; ++ni)
#pragma unroll
      for (int v = 0; v < 8; ++v) {
        const int q = rbase + mi * 16 + v;
        const int k = cbase + ni * 16;
        Sb[(size_t)q * SEQ + k] = acc[mi][ni][v];   // k>q entries are never read
      }
}

// ---- Kernel 3: causal softmax row-wise, P[bf16] with zeros above diagonal ---
__global__ __launch_bounds__(256)
void softmax_kernel(const float* __restrict__ Sc, bf16_t* __restrict__ P) {
  const int row = blockIdx.x;                   // 0..8191
  const int b = row >> 11;
  const int q = row & (SEQ - 1);
  const float* s = Sc + (size_t)b * SEQ * SEQ + (size_t)q * SEQ;
  bf16_t*      p = P  + (size_t)b * SEQ * SEQ + (size_t)q * SEQ;
  const int n   = q + 1;                        // causal prefix length
  const int tid = threadIdx.x;
  __shared__ float red[8];

  float m = -3.402823466e38f;
  for (int j = tid; j < n; j += 256) m = fmaxf(m, s[j]);
#pragma unroll
  for (int o = 16; o > 0; o >>= 1) m = fmaxf(m, __shfl_xor(m, o, 32));
  if ((tid & 31) == 0) red[tid >> 5] = m;
  __syncthreads();
#pragma unroll
  for (int w = 0; w < 8; ++w) m = fmaxf(m, red[w]);
  __syncthreads();

  float sum = 0.0f;
  for (int j = tid; j < n; j += 256) sum += __expf(s[j] - m);
#pragma unroll
  for (int o = 16; o > 0; o >>= 1) sum += __shfl_xor(sum, o, 32);
  if ((tid & 31) == 0) red[tid >> 5] = sum;
  __syncthreads();
  sum = 0.0f;
#pragma unroll
  for (int w = 0; w < 8; ++w) sum += red[w];
  const float inv = 1.0f / sum;

  for (int j = tid; j < SEQ; j += 256)
    p[j] = (j < n) ? (bf16_t)(__expf(s[j] - m) * inv) : (bf16_t)0.0f;
}

// ---- Kernel 4: O[f32] = P @ V; block 64(q)x128(n); P tiles pipelined via TDM
__global__ __launch_bounds__(128)
void pv_gemm_kernel(const bf16_t* __restrict__ P, const bf16_t* __restrict__ V,
                    float* __restrict__ O) {
  __shared__ bf16_t sA[2][64 * LDT];            // P tiles, TDM-padded to 80B rows
  __shared__ bf16_t sB[2][128 * LDT];           // V^T: 128(n) x 32(k)
  const int b  = blockIdx.z;
  const int q0 = blockIdx.y * 64;
  const int n0 = blockIdx.x * 128;
  const int tid  = threadIdx.x;
  const int lane = tid & 31;
  const int wave = tid >> 5;
  const bf16_t* Pb = P + (size_t)b * SEQ * SEQ;
  const bf16_t* Vb = V + (size_t)b * SEQ * DMODEL;
  float*        Ob = O + (size_t)b * SEQ * DMODEL;
  const uint32_t sA_off  = (uint32_t)(uintptr_t)(&sA[0][0]);
  const uint32_t sA_size = (uint32_t)(64 * LDT * sizeof(bf16_t));

  const int kmax = q0 + 64;                     // P is zero for k > q
  // prologue: TDM for step 0 + stage V-tile 0 into registers
  if (wave == 0)
    tdm_load_tile_2d(sA_off, Pb + (size_t)q0 * SEQ, 64, SEQ, SEQ, SEQ);
  v8bf regB[4];
#pragma unroll
  for (int i = 0; i < 4; ++i) {
    const int idx = tid + i * 128;
    const int r = idx >> 4, c8 = idx & 15;
    regB[i] = *(const v8bf*)(Vb + (size_t)r * DMODEL + n0 + c8 * 8);
  }

  v8f acc[4][2] = {};
  for (int k0 = 0, step = 0; k0 < kmax; k0 += BK, ++step) {
    const int cur = step & 1;
    // commit staged V tile into sB[cur] (transposed)
#pragma unroll
    for (int i = 0; i < 4; ++i) {
      const int idx = tid + i * 128;
      const int r = idx >> 4, c8 = idx & 15;
#pragma unroll
      for (int j = 0; j < 8; ++j) sB[cur][(c8 * 8 + j) * LDT + r] = regB[i][j];
    }
    if (wave == 0) __builtin_amdgcn_s_wait_tensorcnt(0);  // P tile `cur` landed
    __syncthreads();
    // kick off next step's async work: it targets the *other* buffers, which
    // every wave finished reading before the barrier above
    if (k0 + BK < kmax) {
      if (wave == 0)
        tdm_load_tile_2d(sA_off + (cur ^ 1) * sA_size,
                         Pb + (size_t)q0 * SEQ + (k0 + BK), 64, SEQ, SEQ, SEQ);
#pragma unroll
      for (int i = 0; i < 4; ++i) {
        const int idx = tid + i * 128;
        const int r = idx >> 4, c8 = idx & 15;
        regB[i] = *(const v8bf*)(Vb + (size_t)(k0 + BK + r) * DMODEL + n0 + c8 * 8);
      }
    }
    v16bf af[4], bfv[2];
#pragma unroll
    for (int mi = 0; mi < 4; ++mi) af[mi] = load_frag_a(&sA[cur][(mi * 16) * LDT], LDT, lane);
#pragma unroll
    for (int ni = 0; ni < 2; ++ni)
      bfv[ni] = load_frag_b(&sB[cur][(wave * 32 + ni * 16) * LDT], LDT, lane);
#pragma unroll
    for (int mi = 0; mi < 4; ++mi)
#pragma unroll
      for (int ni = 0; ni < 2; ++ni)
        acc[mi][ni] = wmma_bf16(af[mi], bfv[ni], acc[mi][ni]);
    // no trailing barrier: double-buffered
  }

  const int rbase = q0 + ((lane & 16) ? 8 : 0);
  const int cbase = n0 + wave * 32 + (lane & 15);
#pragma unroll
  for (int mi = 0; mi < 4; ++mi)
#pragma unroll
    for (int ni = 0; ni < 2; ++ni)
#pragma unroll
      for (int v = 0; v < 8; ++v) {
        const int row = rbase + mi * 16 + v;
        const int col = cbase + ni * 16;
        Ob[(size_t)row * DMODEL + col] = acc[mi][ni][v];
      }
}

// ---- Host orchestration -----------------------------------------------------
extern "C" void kernel_launch(void* const* d_in, const int* in_sizes, int n_in,
                              void* d_out, int out_size, void* d_ws, size_t ws_size,
                              hipStream_t stream) {
  (void)in_sizes; (void)n_in; (void)out_size; (void)ws_size;
  const float* x  = (const float*)d_in[0];
  const float* Wq = (const float*)d_in[1];
  const float* Wk = (const float*)d_in[2];
  const float* Wv = (const float*)d_in[3];
  float* out = (float*)d_out;

  char* w = (char*)d_ws;
  const size_t qkv_bytes = (size_t)ROWS * DMODEL * sizeof(bf16_t);       // 16 MB each
  const size_t sc_bytes  = (size_t)BATCH * SEQ * SEQ * sizeof(float);    // 64 MB
  bf16_t* Qs = (bf16_t*)(w);
  bf16_t* Kb = (bf16_t*)(w + qkv_bytes);
  bf16_t* Vb = (bf16_t*)(w + 2 * qkv_bytes);
  float*  Sc = (float*)(w + 3 * qkv_bytes);
  bf16_t* Pm = (bf16_t*)(w + 3 * qkv_bytes + sc_bytes);                  // 32 MB

  const dim3 gproj(DMODEL / 128, ROWS / 64);                             // 8 x 128
  qkv_gemm_kernel<<<gproj, 128, 0, stream>>>(x, Wq, Qs, 1.0f / 32.0f);   // Q pre-scaled
  qkv_gemm_kernel<<<gproj, 128, 0, stream>>>(x, Wk, Kb, 1.0f);
  qkv_gemm_kernel<<<gproj, 128, 0, stream>>>(x, Wv, Vb, 1.0f);
  scores_kernel<<<dim3(SEQ / 128, SEQ / 64, BATCH), 128, 0, stream>>>(Qs, Kb, Sc);
  softmax_kernel<<<ROWS, 256, 0, stream>>>(Sc, Pm);
  pv_gemm_kernel<<<dim3(DMODEL / 128, SEQ / 64, BATCH), 128, 0, stream>>>(Pm, Vb, out);
}
